// Block_54408645706395
// MI455X (gfx1250) — matrix operand order repaired and checked by reading, not compiled, not run on hardware
//
#include <hip/hip_runtime.h>
#include <hip/hip_bf16.h>
#include <math.h>

// ---------------- problem constants ----------------
#define BATCH   8
#define HH      64
#define WW      64
#define NTOK    (HH*WW)          // 4096
#define CDIM    384
#define HEADS   8
#define HD      48
#define HDP     64               // head_dim padded to multiple of 32 for WMMA K
#define NK      256              // (64/4)^2 downsampled kv tokens
#define HID     1536
#define BN      (BATCH*NTOK)     // 32768
#define BK      (BATCH*NK)       // 2048
#define SCALE_F 0.14433756729740643f   // 48^-0.5

#define FT      16               // fused-MLP tokens per block
#define LDSROW  (HID + 8)        // 1544 halfs: +16B pad -> bank-spread ds_load_b128

typedef __attribute__((ext_vector_type(16))) _Float16 v16h;
typedef __attribute__((ext_vector_type(8)))  _Float16 v8h;
typedef __attribute__((ext_vector_type(8)))  float    v8f;

// -------- WMMA operand loaders (CDNA5 16-bit A/B VGPR layouts, wave32) --------
// A 16x32 f16: lanes 0-15 hold K=0-7 & 16-23, lanes 16-31 hold K=8-15 & 24-31.
__device__ inline v16h load_a16(const _Float16* __restrict__ p, int lane) {
  const int klo = (lane & 16) ? 8 : 0;
  union { v16h v; v8h h[2]; } u;
  u.h[0] = *reinterpret_cast<const v8h*>(p + klo);
  u.h[1] = *reinterpret_cast<const v8h*>(p + klo + 16);
  return u.v;
}
// LDS flavor (ds_load_b128 path for the fused MLP)
__device__ inline v16h load_a16_lds(const _Float16* p, int lane) {
  const int klo = (lane & 16) ? 8 : 0;
  union { v16h v; v8h h[2]; } u;
  u.h[0] = *reinterpret_cast<const v8h*>(p + klo);
  u.h[1] = *reinterpret_cast<const v8h*>(p + klo + 16);
  return u.v;
}
// B 32x16 f16 (pre-transposed [N x K] rows): lane holds column n=lane&15;
// lanes 0-15 carry K=0-7 & 16-23, lanes 16-31 carry K=8-15 & 24-31 (matches A).
__device__ inline v16h load_b16(const _Float16* __restrict__ p, int lane) {
  const int klo = (lane & 16) ? 8 : 0;
  union { v16h v; v8h h[2]; } u;
  u.h[0] = *reinterpret_cast<const v8h*>(p + klo);
  u.h[1] = *reinterpret_cast<const v8h*>(p + klo + 16);
  return u.v;
}

__device__ inline v8f wmma_f16(v16h a, v16h b, v8f c) {
  return __builtin_amdgcn_wmma_f32_16x16x32_f16(false, a, false, b, (short)0, c, false, false);
}

__device__ inline float gelu_exact(float x) {
  return 0.5f * x * (1.0f + erff(x * 0.70710678118654752f));
}

// ---------------- weight fp32 [K x N] -> f16 transposed [N x K] ----------------
__global__ __launch_bounds__(256)
void wtrans_kernel(const float* __restrict__ w, _Float16* __restrict__ wt, int K, int N) {
  size_t idx = (size_t)blockIdx.x * blockDim.x + threadIdx.x;   // over N*K, dst-linear
  if (idx >= (size_t)K * N) return;
  int k = (int)(idx % K);
  int n = (int)(idx / K);
  wt[idx] = (_Float16)w[(size_t)k * N + n];
}

// ---------------- CPE: depthwise 3x3 SAME conv + bias + residual ----------------
__global__ __launch_bounds__(256)
void cpe_kernel(const float* __restrict__ x, const float* __restrict__ w,
                const float* __restrict__ b, float* __restrict__ x1) {
  size_t idx = (size_t)blockIdx.x * blockDim.x + threadIdx.x;   // B*N*C
  int c = (int)(idx % CDIM);
  size_t t = idx / CDIM;
  int px = (int)(t % WW);  t /= WW;
  int py = (int)(t % HH);
  int bb = (int)(t / HH);
  float acc = b[c];
  #pragma unroll
  for (int dy = -1; dy <= 1; ++dy) {
    #pragma unroll
    for (int dx = -1; dx <= 1; ++dx) {
      int yy = py + dy, xx = px + dx;
      if (yy >= 0 && yy < HH && xx >= 0 && xx < WW)
        acc += x[(((size_t)bb * HH + yy) * WW + xx) * CDIM + c] *
               w[((dy + 1) * 3 + (dx + 1)) * CDIM + c];
    }
  }
  x1[idx] = x[idx] + acc;
}

// ---------------- LayerNorm over C=384, wave-per-token, f16 output ----------------
__global__ __launch_bounds__(256)
void ln_kernel(const float* __restrict__ x, const float* __restrict__ g,
               const float* __restrict__ be, _Float16* __restrict__ outh, int ntok) {
  int lane = threadIdx.x & 31;
  int wid  = threadIdx.x >> 5;
  int tok  = blockIdx.x * 8 + wid;
  if (tok >= ntok) return;
  const float* row = x + (size_t)tok * CDIM;
  float vals[CDIM / 32];
  float s = 0.f, s2 = 0.f;
  #pragma unroll
  for (int i = 0; i < CDIM / 32; ++i) {
    float v = row[lane + 32 * i];
    vals[i] = v; s += v; s2 += v * v;
  }
  #pragma unroll
  for (int o = 16; o > 0; o >>= 1) {       // wave32 tree reduce
    s  += __shfl_xor(s,  o, 32);
    s2 += __shfl_xor(s2, o, 32);
  }
  float mean = s * (1.0f / CDIM);
  float var  = s2 * (1.0f / CDIM) - mean * mean;
  float inv  = rsqrtf(var + 1e-5f);
  #pragma unroll
  for (int i = 0; i < CDIM / 32; ++i) {
    int c = lane + 32 * i;
    outh[(size_t)tok * CDIM + c] = (_Float16)((vals[i] - mean) * inv * g[c] + be[c]);
  }
}

// ---------------- KV downsample: depthwise 4x4 stride-4 VALID, f16 in/out ----------------
__global__ __launch_bounds__(256)
void kvconv_kernel(const _Float16* __restrict__ h, const float* __restrict__ w,
                   _Float16* __restrict__ kvimg) {
  size_t idx = (size_t)blockIdx.x * blockDim.x + threadIdx.x;   // B*256*C
  int c = (int)(idx % CDIM);
  size_t t = idx / CDIM;
  int ox = (int)(t % 16); t /= 16;
  int oy = (int)(t % 16);
  int bb = (int)(t / 16);
  float acc = 0.f;
  #pragma unroll
  for (int ky = 0; ky < 4; ++ky)
    #pragma unroll
    for (int kx = 0; kx < 4; ++kx)
      acc += (float)h[(((size_t)bb * HH + oy * 4 + ky) * WW + ox * 4 + kx) * CDIM + c] *
             w[(ky * 4 + kx) * CDIM + c];
  kvimg[idx] = (_Float16)acc;
}

// ---------------- epilogue helper for one 16x16 C/D tile ----------------
__device__ inline void store_tile(v8f acc, int mBase, int n, int N,
                                  const float* __restrict__ bias,
                                  const float* __restrict__ residual,
                                  float* __restrict__ Cf, _Float16* __restrict__ Ch,
                                  int do_gelu, int qscatter, int lane) {
  const int mTop = mBase + ((lane & 16) ? 8 : 0);
  const float bb = bias ? bias[n] : 0.f;
  int cc = n, ld = N;
  if (qscatter) { cc = (n / HD) * HDP + (n % HD); ld = HEADS * HDP; }
  #pragma unroll
  for (int r = 0; r < 8; ++r) {
    int m = mTop + r;
    float v = acc[r] + bb;
    if (do_gelu) v = gelu_exact(v);
    size_t ir = (size_t)m * N + n;
    if (residual) v += residual[ir];
    if (Cf) Cf[ir] = v;
    if (Ch) Ch[(size_t)m * ld + cc] = (_Float16)v;
  }
}

// -------- main WMMA GEMM: C[M,N] = A[M,K] @ Bt[N,K]^T ; 32x32 per wave, 128x64 per block --------
__global__ __launch_bounds__(256)
void gemm_wmma_f16(const _Float16* __restrict__ A, const _Float16* __restrict__ Bt,
                   const float* __restrict__ bias, const float* __restrict__ residual,
                   float* __restrict__ Cf, _Float16* __restrict__ Ch,
                   int M, int K, int N, int do_gelu, int qscatter) {
  const int lane = threadIdx.x & 31;
  const int wid  = threadIdx.x >> 5;            // 8 waves
  const int wm   = wid & 3, wn = wid >> 2;      // 4 (M) x 2 (N)
  const int mBase = blockIdx.y * 128 + wm * 32;
  const int nBase = blockIdx.x * 64 + wn * 32;
  const int l15 = lane & 15;
  const _Float16* arow0 = A  + (size_t)(mBase +      l15) * K;
  const _Float16* arow1 = A  + (size_t)(mBase + 16 + l15) * K;
  const _Float16* brow0 = Bt + (size_t)(nBase +      l15) * K;
  const _Float16* brow1 = Bt + (size_t)(nBase + 16 + l15) * K;
  v8f acc00 = {}, acc01 = {}, acc10 = {}, acc11 = {};
  for (int kk = 0; kk < K; kk += 32) {
    __builtin_prefetch(arow0 + kk + 256, 0, 0);   // global_prefetch_b8
    __builtin_prefetch(arow1 + kk + 256, 0, 0);
    v16h a0 = load_a16(arow0 + kk, lane);
    v16h a1 = load_a16(arow1 + kk, lane);
    v16h b0 = load_b16(brow0 + kk, lane);
    v16h b1 = load_b16(brow1 + kk, lane);
    acc00 = wmma_f16(a0, b0, acc00);
    acc01 = wmma_f16(a0, b1, acc01);
    acc10 = wmma_f16(a1, b0, acc10);
    acc11 = wmma_f16(a1, b1, acc11);
  }
  store_tile(acc00, mBase,      nBase +      l15, N, bias, residual, Cf, Ch, do_gelu, qscatter, lane);
  store_tile(acc01, mBase,      nBase + 16 + l15, N, bias, residual, Cf, Ch, do_gelu, qscatter, lane);
  store_tile(acc10, mBase + 16, nBase +      l15, N, bias, residual, Cf, Ch, do_gelu, qscatter, lane);
  store_tile(acc11, mBase + 16, nBase + 16 + l15, N, bias, residual, Cf, Ch, do_gelu, qscatter, lane);
}

// ---------------- zero the pad columns (d=48..63) of padded q ----------------
__global__ __launch_bounds__(256)
void qzero_kernel(_Float16* __restrict__ qp) {
  size_t idx = (size_t)blockIdx.x * blockDim.x + threadIdx.x;   // BN*HEADS*16
  int d = (int)(idx & 15);
  size_t t = idx >> 4;
  int h = (int)(t & 7);
  size_t tok = t >> 3;
  qp[tok * (HEADS * HDP) + h * HDP + HD + d] = (_Float16)0.0f;
}

// ---------------- softmax over key tokens (axis Nk) per (b, channel), in place ----------------
__global__ __launch_bounds__(256)
void softmax_kernel(float* __restrict__ k) {
  int c = blockIdx.x % CDIM;
  int b = blockIdx.x / CDIM;
  int n = threadIdx.x;                 // 0..255 = key index
  __shared__ float red[256];
  size_t idx = ((size_t)b * NK + n) * CDIM + c;
  float v = k[idx];
  red[n] = v; __syncthreads();
  for (int s = 128; s > 0; s >>= 1) { if (n < s) red[n] = fmaxf(red[n], red[n + s]); __syncthreads(); }
  float mx = red[0]; __syncthreads();
  float e = expf(v - mx);
  red[n] = e; __syncthreads();
  for (int s = 128; s > 0; s >>= 1) { if (n < s) red[n] += red[n + s]; __syncthreads(); }
  k[idx] = e / red[0];
}

// ---------------- kv = SCALE * ksm^T @ v per (b,h); Bt-layout f16 [vd][kd pad64] ----------------
__global__ __launch_bounds__(256)
void kvmat_kernel(const float* __restrict__ ksm, const float* __restrict__ v,
                  _Float16* __restrict__ kvT) {
  size_t idx = (size_t)blockIdx.x * blockDim.x + threadIdx.x;   // B*H*48*64
  int kd = (int)(idx & 63);
  size_t t = idx >> 6;
  int vd = (int)(t % HD); t /= HD;
  int h  = (int)(t & 7);
  int b  = (int)(t >> 3);
  float acc = 0.f;
  if (kd < HD) {
    int ck = h * HD + kd, cv = h * HD + vd;
    for (int n = 0; n < NK; ++n) {
      size_t base = ((size_t)b * NK + n) * CDIM;
      acc += ksm[base + ck] * v[base + cv];
    }
    acc *= SCALE_F;
  }
  kvT[idx] = (_Float16)acc;
}

// ---------------- attn = q_pad @ kv per (b,h), WMMA K=64, N=48 (3 tiles), f16 out ----------------
__global__ __launch_bounds__(256)
void attn_wmma(const _Float16* __restrict__ qp,   // [B*N, HEADS*64]
               const _Float16* __restrict__ kvT,  // [B*HEADS][48][64]
               _Float16* __restrict__ attn) {     // [B*N, 384]
  int blk = blockIdx.x;                // B*HEADS*32 blocks; 128 token rows per block
  int mchunk = blk & 31;
  int bh = blk >> 5;
  int h = bh & 7, b = bh >> 3;
  int lane = threadIdx.x & 31;
  int wid  = threadIdx.x >> 5;
  int mBase = mchunk * 128 + wid * 16;
  const _Float16* arow = qp + ((size_t)b * NTOK + mBase + (lane & 15)) * (HEADS * HDP) + h * HDP;
  const _Float16* kvb  = kvT + (size_t)bh * HD * HDP;
  v8f acc[3] = {{}, {}, {}};
  #pragma unroll
  for (int kk = 0; kk < HDP; kk += 32) {
    v16h a = load_a16(arow + kk, lane);
    #pragma unroll
    for (int nt = 0; nt < 3; ++nt) {
      v16h bm = load_b16(kvb + (size_t)(nt * 16 + (lane & 15)) * HDP + kk, lane);
      acc[nt] = wmma_f16(a, bm, acc[nt]);
    }
  }
  int mTop = mBase + ((lane & 16) ? 8 : 0);
  #pragma unroll
  for (int nt = 0; nt < 3; ++nt) {
    int col = h * HD + nt * 16 + (lane & 15);
    #pragma unroll
    for (int r = 0; r < 8; ++r) {
      size_t idx = ((size_t)b * NTOK + mTop + r) * CDIM + col;
      attn[idx] = (_Float16)acc[nt][r];
    }
  }
}

// -------- fused MLP: out = gelu(A@W1+b1)@W2 + b2 + residual; hidden lives in LDS --------
__global__ __launch_bounds__(256)
void mlp_fused(const _Float16* __restrict__ A,    // h2 f16 [BN x 384]
               const _Float16* __restrict__ w1t,  // [1536 x 384]
               const float* __restrict__ b1,
               const _Float16* __restrict__ w2t,  // [384 x 1536]
               const float* __restrict__ b2,
               const float* __restrict__ residual,// x2
               float* __restrict__ out) {
  extern __shared__ _Float16 hsh[];               // FT x LDSROW halfs (48KB)
  const int lane = threadIdx.x & 31;
  const int wid  = threadIdx.x >> 5;              // 8 waves, all on N (32 cols each)
  const int l15  = lane & 15;
  const int tok0 = blockIdx.x * FT;
  const _Float16* arow = A + (size_t)(tok0 + l15) * CDIM;

  // GEMM1: 16 x 1536, 256 cols per pass, gelu -> LDS
  for (int nc = 0; nc < HID; nc += 256) {
    const int n0 = nc + wid * 32;
    const _Float16* brow0 = w1t + (size_t)(n0 +      l15) * CDIM;
    const _Float16* brow1 = w1t + (size_t)(n0 + 16 + l15) * CDIM;
    v8f acc0 = {}, acc1 = {};
    #pragma unroll
    for (int kk = 0; kk < CDIM; kk += 32) {
      v16h a  = load_a16(arow  + kk, lane);
      v16h b0 = load_b16(brow0 + kk, lane);
      v16h b1v = load_b16(brow1 + kk, lane);
      acc0 = wmma_f16(a, b0,  acc0);
      acc1 = wmma_f16(a, b1v, acc1);
    }
    const int c0 = n0 + l15, c1 = c0 + 16;
    const int mTop = (lane & 16) ? 8 : 0;
    const float bb0 = b1[c0], bb1 = b1[c1];
    #pragma unroll
    for (int r = 0; r < 8; ++r) {
      int m = mTop + r;
      hsh[m * LDSROW + c0] = (_Float16)gelu_exact(acc0[r] + bb0);
      hsh[m * LDSROW + c1] = (_Float16)gelu_exact(acc1[r] + bb1);
    }
  }
  __syncthreads();

  // GEMM2: 16 x 384, A from LDS (ds_load_b128), K=1536
  for (int nc = 0; nc < CDIM; nc += 256) {
    const int n0 = nc + wid * 32;
    if (n0 >= CDIM) continue;                      // wave-uniform
    const _Float16* brow0 = w2t + (size_t)(n0 +      l15) * HID;
    const _Float16* brow1 = w2t + (size_t)(n0 + 16 + l15) * HID;
    const _Float16* alds  = hsh + (size_t)l15 * LDSROW;
    v8f acc0 = {}, acc1 = {};
    for (int kk = 0; kk < HID; kk += 32) {
      __builtin_prefetch(brow0 + kk + 256, 0, 0);
      v16h a  = load_a16_lds(alds + kk, lane);
      v16h b0 = load_b16(brow0 + kk, lane);
      v16h b1v = load_b16(brow1 + kk, lane);
      acc0 = wmma_f16(a, b0,  acc0);
      acc1 = wmma_f16(a, b1v, acc1);
    }
    const int c0 = n0 + l15, c1 = c0 + 16;
    const int mTop = tok0 + ((lane & 16) ? 8 : 0);
    const float bb0 = b2[c0], bb1 = b2[c1];
    #pragma unroll
    for (int r = 0; r < 8; ++r) {
      size_t i0 = (size_t)(mTop + r) * CDIM + c0;
      size_t i1 = (size_t)(mTop + r) * CDIM + c1;
      out[i0] = acc0[r] + bb0 + residual[i0];
      out[i1] = acc1[r] + bb1 + residual[i1];
    }
  }
}

// =========================================================================
extern "C" void kernel_launch(void* const* d_in, const int* in_sizes, int n_in,
                              void* d_out, int out_size, void* d_ws, size_t ws_size,
                              hipStream_t stream) {
  const float* x      = (const float*)d_in[0];
  const float* cpe_w  = (const float*)d_in[1];
  const float* cpe_b  = (const float*)d_in[2];
  const float* q_w    = (const float*)d_in[3];
  const float* q_b    = (const float*)d_in[4];
  const float* kv_w   = (const float*)d_in[5];
  const float* k_w    = (const float*)d_in[6];
  const float* v_w    = (const float*)d_in[7];
  const float* v_b    = (const float*)d_in[8];
  const float* proj_w = (const float*)d_in[9];
  const float* proj_b = (const float*)d_in[10];
  const float* ln1_g  = (const float*)d_in[11];
  const float* ln1_b  = (const float*)d_in[12];
  const float* ln2_g  = (const float*)d_in[13];
  const float* ln2_b  = (const float*)d_in[14];
  const float* w1     = (const float*)d_in[15];
  const float* b1     = (const float*)d_in[16];
  const float* w2     = (const float*)d_in[17];
  const float* b2     = (const float*)d_in[18];
  float* out = (float*)d_out;

  // -------- workspace carve-up (256B aligned) --------
  size_t off = 0;
  char* base = (char*)d_ws;
  auto alloc = [&](size_t bytes) -> void* {
    void* p = base + off;
    off = (off + bytes + 255) & ~(size_t)255;
    return p;
  };
  _Float16* qWt    = (_Float16*)alloc((size_t)CDIM * CDIM * 2);
  _Float16* kWt    = (_Float16*)alloc((size_t)CDIM * CDIM * 2);
  _Float16* vWt    = (_Float16*)alloc((size_t)CDIM * CDIM * 2);
  _Float16* pWt    = (_Float16*)alloc((size_t)CDIM * CDIM * 2);
  _Float16* w1t    = (_Float16*)alloc((size_t)HID * CDIM * 2);
  _Float16* w2t    = (_Float16*)alloc((size_t)CDIM * HID * 2);
  float*    x1     = (float*)   alloc((size_t)BN * CDIM * 4);
  _Float16* h16    = (_Float16*)alloc((size_t)BN * CDIM * 2);
  _Float16* kv16   = (_Float16*)alloc((size_t)BK * CDIM * 2);
  _Float16* qp16   = (_Float16*)alloc((size_t)BN * HEADS * HDP * 2);
  float*    kbuf   = (float*)   alloc((size_t)BK * CDIM * 4);
  float*    vbuf   = (float*)   alloc((size_t)BK * CDIM * 4);
  _Float16* kvT    = (_Float16*)alloc((size_t)BATCH * HEADS * HD * HDP * 2);
  _Float16* attn16 = (_Float16*)alloc((size_t)BN * CDIM * 2);
  float*    x2     = (float*)   alloc((size_t)BN * CDIM * 4);
  _Float16* h2_16  = (_Float16*)alloc((size_t)BN * CDIM * 2);
  if (off > ws_size) return;   // workspace too small -> no-op

  // -------- weights -> f16 transposed [N x K] --------
  wtrans_kernel<<<(CDIM * CDIM + 255) / 256, 256, 0, stream>>>(q_w,    qWt, CDIM, CDIM);
  wtrans_kernel<<<(CDIM * CDIM + 255) / 256, 256, 0, stream>>>(k_w,    kWt, CDIM, CDIM);
  wtrans_kernel<<<(CDIM * CDIM + 255) / 256, 256, 0, stream>>>(v_w,    vWt, CDIM, CDIM);
  wtrans_kernel<<<(CDIM * CDIM + 255) / 256, 256, 0, stream>>>(proj_w, pWt, CDIM, CDIM);
  wtrans_kernel<<<(CDIM * HID  + 255) / 256, 256, 0, stream>>>(w1,     w1t, CDIM, HID);
  wtrans_kernel<<<(HID  * CDIM + 255) / 256, 256, 0, stream>>>(w2,     w2t, HID,  CDIM);

  // -------- CPE + residual --------
  cpe_kernel<<<(size_t)BN * CDIM / 256, 256, 0, stream>>>(x, cpe_w, cpe_b, x1);

  // -------- LN1 -> h (f16) --------
  ln_kernel<<<BN / 8, 256, 0, stream>>>(x1, ln1_g, ln1_b, h16, BN);

  // -------- kv downsample conv --------
  kvconv_kernel<<<(size_t)BK * CDIM / 256, 256, 0, stream>>>(h16, kv_w, kv16);

  // -------- q / k / v projections (WMMA); q scatters into padded head layout --------
  qzero_kernel<<<(size_t)BN * HEADS * 16 / 256, 256, 0, stream>>>(qp16);
  gemm_wmma_f16<<<dim3(CDIM / 64, BN / 128), 256, 0, stream>>>(
      h16, qWt, q_b, nullptr, nullptr, qp16, BN, CDIM, CDIM, 0, 1);
  gemm_wmma_f16<<<dim3(CDIM / 64, BK / 128), 256, 0, stream>>>(
      kv16, kWt, nullptr, nullptr, kbuf, nullptr, BK, CDIM, CDIM, 0, 0);
  gemm_wmma_f16<<<dim3(CDIM / 64, BK / 128), 256, 0, stream>>>(
      kv16, vWt, v_b, nullptr, vbuf, nullptr, BK, CDIM, CDIM, 0, 0);

  // -------- softmax over keys, kv matrix (scale folded in) --------
  softmax_kernel<<<BATCH * CDIM, 256, 0, stream>>>(kbuf);
  kvmat_kernel<<<(BATCH * HEADS * HD * HDP) / 256, 256, 0, stream>>>(kbuf, vbuf, kvT);

  // -------- attn = q @ kv (WMMA) --------
  attn_wmma<<<BATCH * HEADS * (NTOK / 128), 256, 0, stream>>>(qp16, kvT, attn16);

  // -------- proj + residual -> x2 --------
  gemm_wmma_f16<<<dim3(CDIM / 64, BN / 128), 256, 0, stream>>>(
      attn16, pWt, proj_b, x1, x2, nullptr, BN, CDIM, CDIM, 0, 0);

  // -------- LN2 -> h2 (f16) --------
  ln_kernel<<<BN / 8, 256, 0, stream>>>(x2, ln2_g, ln2_b, h2_16, BN);

  // -------- fused MLP (hidden stays in LDS) -> out --------
  mlp_fused<<<BN / FT, 256, (size_t)FT * LDSROW * sizeof(_Float16), stream>>>(
      h2_16, w1t, b1, w2t, b2, x2, out);
}
